// PureSSMBlock_55576876811018
// MI455X (gfx1250) — matrix operand order repaired and checked by reading, not compiled, not run on hardware
//
#include <hip/hip_runtime.h>

// ---------------------------------------------------------------------------
// Problem constants (match reference)
// ---------------------------------------------------------------------------
#define BB      4
#define LL      1024
#define TOK     (BB * LL)          // 4096 tokens
#define DMODEL  768
#define DINNER  1536
#define DSTATE  16
#define DCONV   4
#define DTRANK  48
#define NEXP    4
#define DFF     3072
#define EPSV    1e-6f

typedef __attribute__((ext_vector_type(16))) __bf16 v16bf;
typedef __attribute__((ext_vector_type(8)))  float  v8f;

union BF16Frag {
    v16bf v;
    uint4 q[2];
    unsigned short u[16];
};

__device__ __forceinline__ unsigned short f32_to_bf16(float f) {
    unsigned int x = __float_as_uint(f);
    unsigned int r = x + 0x7FFFu + ((x >> 16) & 1u);   // RNE
    return (unsigned short)(r >> 16);
}
__device__ __forceinline__ unsigned int pack_bf16x2(float a, float b) {
    return (unsigned int)f32_to_bf16(a) | ((unsigned int)f32_to_bf16(b) << 16);
}
__device__ __forceinline__ float siluf(float x) { return x / (1.f + __expf(-x)); }
__device__ __forceinline__ float softplusf(float x) {
    return fmaxf(x, 0.f) + log1pf(__expf(-fabsf(x)));
}

// ---------------------------------------------------------------------------
// Unified bf16-WMMA GEMM:  C[M,N] = epilog(A[M,K] @ B[K,N])
//   - 128 threads (4 waves); block tile 32(M) x 64(N), K-step 32
//   - wave w: rows (w>>1)*16, cols (w&1)*32 (two 16x16 WMMAs share A frag)
//   - A staged row-major bf16 (stride 40 halves = 80B, 16B aligned)
//   - B staged TRANSPOSED [n][k] so each lane's 16 K-values are two 16B runs
//   - optional per-expert bucketing (counts/offsets/perm), gather-A/scatter-C,
//     bias+softplus, dual-B silu-gate, residual + layer-scale + row-scale.
// Requires: K % 4 == 0, N % 4 == 0, ldb % 4 == 0 (true for all uses here).
// ---------------------------------------------------------------------------
#define LDSS 40   // LDS row stride in halves (80 bytes, 16B aligned)

__global__ __launch_bounds__(128)
void wmma_gemm_kernel(const float* __restrict__ A, const float* __restrict__ B0,
                      const float* __restrict__ B1, float* __restrict__ C,
                      int M, int N, int K, int lda, int ldb, int ldc,
                      const int* __restrict__ counts, const int* __restrict__ offsets,
                      const int* __restrict__ perm, int gatherA, int scatterC,
                      long expert_stride,
                      const float* __restrict__ bias, int act,
                      const float* __restrict__ resid, int ldres,
                      const float* __restrict__ lscale,
                      const float* __restrict__ rowscale)
{
    __shared__ __attribute__((aligned(16))) unsigned short sA [32 * LDSS];
    __shared__ __attribute__((aligned(16))) unsigned short sB0[64 * LDSS];  // [n][k]
    __shared__ __attribute__((aligned(16))) unsigned short sB1[64 * LDSS];  // [n][k]

    const int e = blockIdx.z;
    int mcnt = M, roff = 0;
    if (counts) { mcnt = counts[e]; roff = offsets[e]; }
    const int mt = blockIdx.y;
    if (mt * 32 >= mcnt) return;
    const int nt = blockIdx.x;

    const float* Bp0 = B0 + (long)e * expert_stride;
    const float* Bp1 = B1 ? (B1 + (long)e * expert_stride) : nullptr;

    const int tid  = threadIdx.x;
    const int wave = tid >> 5;
    const int lane = tid & 31;
    const int wm   = (wave >> 1) * 16;   // wave row offset   (0/16)
    const int wnb  = (wave & 1) * 32;    // wave col offset   (0/32)

    v8f accP0 = {}, accP1 = {}, accQ0 = {}, accQ1 = {};

    for (int k0 = 0; k0 < K; k0 += 32) {
        // ---- stage A: 32 rows x 32 k (float4 -> bf16x2 pairs) ----
        #pragma unroll
        for (int it = 0; it < 2; ++it) {
            int idx4 = it * 128 + tid;          // 0..255
            int r    = idx4 >> 3;               // 0..31
            int c4   = (idx4 & 7) * 4;          // 0..28
            float4 v = make_float4(0.f, 0.f, 0.f, 0.f);
            int grow = mt * 32 + r;
            if (grow < mcnt && (k0 + c4) < K) {
                int arow = gatherA ? perm[roff + grow] : (roff + grow);
                v = *(const float4*)&A[(long)arow * lda + (k0 + c4)];
                if (k0 + 32 < K)
                    __builtin_prefetch(&A[(long)arow * lda + (k0 + 32 + c4)], 0, 1);
            }
            *(uint2*)&sA[r * LDSS + c4] =
                make_uint2(pack_bf16x2(v.x, v.y), pack_bf16x2(v.z, v.w));
        }
        // ---- stage B: 32 k x 64 n, transposed into [n][k] ----
        #pragma unroll
        for (int it = 0; it < 4; ++it) {
            int idx4 = it * 128 + tid;          // 0..511
            int r    = idx4 >> 4;               // k row 0..31
            int c4   = (idx4 & 15) * 4;         // n col 0..60
            int gn   = nt * 64 + c4;
            bool ok  = ((k0 + r) < K) && (gn < N);
            float4 v0 = make_float4(0.f, 0.f, 0.f, 0.f);
            if (ok) {
                v0 = *(const float4*)&Bp0[(long)(k0 + r) * ldb + gn];
                if (k0 + 32 < K)
                    __builtin_prefetch(&Bp0[(long)(k0 + 32 + r) * ldb + gn], 0, 1);
            }
            sB0[(c4 + 0) * LDSS + r] = f32_to_bf16(v0.x);
            sB0[(c4 + 1) * LDSS + r] = f32_to_bf16(v0.y);
            sB0[(c4 + 2) * LDSS + r] = f32_to_bf16(v0.z);
            sB0[(c4 + 3) * LDSS + r] = f32_to_bf16(v0.w);
            if (Bp1) {
                float4 v1 = make_float4(0.f, 0.f, 0.f, 0.f);
                if (ok) v1 = *(const float4*)&Bp1[(long)(k0 + r) * ldb + gn];
                sB1[(c4 + 0) * LDSS + r] = f32_to_bf16(v1.x);
                sB1[(c4 + 1) * LDSS + r] = f32_to_bf16(v1.y);
                sB1[(c4 + 2) * LDSS + r] = f32_to_bf16(v1.z);
                sB1[(c4 + 3) * LDSS + r] = f32_to_bf16(v1.w);
            }
        }
        __syncthreads();

        // ---- fragments: aligned 16B runs -> ds_load_b128 ----
        const int mrow  = wm + (lane & 15);
        const int ahalf = (lane >= 16) ? 8 : 0;      // A: K groups 0..7 / 8..15
        const int bhalf = (lane >= 16) ? 8 : 0;      // B: K groups 0..7 / 8..15
        BF16Frag fa;
        fa.q[0] = *(const uint4*)&sA[mrow * LDSS + ahalf];        // K=ahalf..+7
        fa.q[1] = *(const uint4*)&sA[mrow * LDSS + 16 + ahalf];   // K=16+ahalf..+7

        const int n0 = wnb + (lane & 15);
        BF16Frag fb;
        fb.q[0] = *(const uint4*)&sB0[n0 * LDSS + bhalf];
        fb.q[1] = *(const uint4*)&sB0[n0 * LDSS + 16 + bhalf];
        accP0 = __builtin_amdgcn_wmma_f32_16x16x32_bf16(false, fa.v, false, fb.v,
                                                        (short)0, accP0, false, false);
        fb.q[0] = *(const uint4*)&sB0[(n0 + 16) * LDSS + bhalf];
        fb.q[1] = *(const uint4*)&sB0[(n0 + 16) * LDSS + 16 + bhalf];
        accP1 = __builtin_amdgcn_wmma_f32_16x16x32_bf16(false, fa.v, false, fb.v,
                                                        (short)0, accP1, false, false);
        if (Bp1) {
            fb.q[0] = *(const uint4*)&sB1[n0 * LDSS + bhalf];
            fb.q[1] = *(const uint4*)&sB1[n0 * LDSS + 16 + bhalf];
            accQ0 = __builtin_amdgcn_wmma_f32_16x16x32_bf16(false, fa.v, false, fb.v,
                                                            (short)0, accQ0, false, false);
            fb.q[0] = *(const uint4*)&sB1[(n0 + 16) * LDSS + bhalf];
            fb.q[1] = *(const uint4*)&sB1[(n0 + 16) * LDSS + 16 + bhalf];
            accQ1 = __builtin_amdgcn_wmma_f32_16x16x32_bf16(false, fa.v, false, fb.v,
                                                            (short)0, accQ1, false, false);
        }
        __syncthreads();
    }

    // ---- epilogue; C/D layout: VGPR v -> M = v + (lane>=16 ? 8 : 0) ----
    #pragma unroll
    for (int sub = 0; sub < 2; ++sub) {
        const int ncol = nt * 64 + wnb + sub * 16 + (lane & 15);
        if (ncol >= N) continue;
        #pragma unroll
        for (int v = 0; v < 8; ++v) {
            int r    = wm + v + ((lane >= 16) ? 8 : 0);
            int grow = mt * 32 + r;
            if (grow >= mcnt) continue;
            float a0 = sub ? accP1[v] : accP0[v];
            float val = a0;
            if (act == 1) {                       // softplus(x + bias)
                val = softplusf(a0 + (bias ? bias[ncol] : 0.f));
            } else if (act == 2) {                // silu(c0) * c1
                float a1 = sub ? accQ1[v] : accQ0[v];
                val = siluf(a0) * a1;
            }
            int crow = roff + grow;
            int tok  = scatterC ? perm[crow] : crow;
            if (rowscale) val *= rowscale[tok];
            float outv = val;
            if (resid) outv = resid[(long)tok * ldres + ncol] +
                              (lscale ? lscale[ncol] : 1.f) * val;
            C[(long)(scatterC ? tok : crow) * ldc + ncol] = outv;
        }
    }
}

// ---------------------------------------------------------------------------
// RMSNorm: one block (256 thr) per row of 768
// ---------------------------------------------------------------------------
__global__ __launch_bounds__(256)
void rmsnorm_kernel(const float* __restrict__ x, const float* __restrict__ w,
                    float* __restrict__ out)
{
    __shared__ float red[256];
    const int row = blockIdx.x;
    const int tid = threadIdx.x;
    const float* xr = x + (long)row * DMODEL;
    float ss = 0.f;
    #pragma unroll
    for (int i = 0; i < 3; ++i) {
        float v = xr[tid + i * 256];
        ss += v * v;
    }
    red[tid] = ss;
    __syncthreads();
    for (int s = 128; s > 0; s >>= 1) {
        if (tid < s) red[tid] += red[tid + s];
        __syncthreads();
    }
    float scale = rsqrtf(red[0] / (float)DMODEL + EPSV);
    float* orow = out + (long)row * DMODEL;
    #pragma unroll
    for (int i = 0; i < 3; ++i) {
        int d = tid + i * 256;
        orow[d] = xr[d] * scale * w[d];
    }
}

// ---------------------------------------------------------------------------
// Causal depthwise conv (width 4) + SiLU.  xs = first half of xz rows.
// ---------------------------------------------------------------------------
__global__ __launch_bounds__(256)
void conv_silu_kernel(const float* __restrict__ xz, const float* __restrict__ cw,
                      const float* __restrict__ cb, float* __restrict__ xc)
{
    long idx = (long)blockIdx.x * 256 + threadIdx.x;   // over TOK*DINNER
    int d = (int)(idx % DINNER);
    long tl = idx / DINNER;
    int l = (int)(tl % LL);
    int b = (int)(tl / LL);
    float acc = cb[d];
    #pragma unroll
    for (int j = 0; j < DCONV; ++j) {
        int ls = l - (DCONV - 1) + j;
        if (ls >= 0) {
            float xv = xz[((long)(b * LL + ls)) * (2 * DINNER) + d];
            acc += cw[d * DCONV + j] * xv;
        }
    }
    xc[idx] = siluf(acc);
}

// ---------------------------------------------------------------------------
// Selective scan.  grid (DINNER/256, B); one thread per (b, d) channel.
// Writes ygated = (y + u*D) * silu(z), ready for out_proj.
// ---------------------------------------------------------------------------
__global__ __launch_bounds__(256)
void scan_kernel(const float* __restrict__ delta, const float* __restrict__ dbc,
                 const float* __restrict__ xc, const float* __restrict__ xz,
                 const float* __restrict__ A_log, const float* __restrict__ D_param,
                 float* __restrict__ ygated)
{
    __shared__ float sBC[2 * DSTATE];                  // B then C for current l
    const int tid = threadIdx.x;
    const int d = blockIdx.x * 256 + tid;
    const int b = blockIdx.y;

    float a[DSTATE], h[DSTATE];
    #pragma unroll
    for (int s = 0; s < DSTATE; ++s) {
        a[s] = -__expf(A_log[d * DSTATE + s]);
        h[s] = 0.f;
    }
    const float Dp = D_param[d];

    for (int l = 0; l < LL; ++l) {
        const int t = b * LL + l;
        if (tid < 2 * DSTATE) sBC[tid] = dbc[(long)t * 80 + DTRANK + tid];
        __syncthreads();
        float dlt = delta[(long)t * DINNER + d];
        float u   = xc[(long)t * DINNER + d];
        float du  = dlt * u;
        float y   = 0.f;
        #pragma unroll
        for (int s = 0; s < DSTATE; ++s) {
            h[s] = h[s] * __expf(dlt * a[s]) + du * sBC[s];
            y += h[s] * sBC[DSTATE + s];
        }
        float z = xz[(long)t * (2 * DINNER) + DINNER + d];
        ygated[(long)t * DINNER + d] = (y + u * Dp) * siluf(z);
        __syncthreads();
    }
}

// ---------------------------------------------------------------------------
// Router: logits = hn @ gate_w (768x4); softmax, argmax (first on tie),
// gate, probs; integer-atomic counts (deterministic).
// ---------------------------------------------------------------------------
__global__ __launch_bounds__(128)
void router_kernel(const float* __restrict__ hn, const float* __restrict__ gw,
                   float* __restrict__ probs, float* __restrict__ gate,
                   int* __restrict__ expert, int* __restrict__ counts)
{
    __shared__ float red[NEXP][128];
    const int t = blockIdx.x;
    const int tid = threadIdx.x;
    float acc[NEXP] = {0.f, 0.f, 0.f, 0.f};
    for (int d = tid; d < DMODEL; d += 128) {
        float xv = hn[(long)t * DMODEL + d];
        #pragma unroll
        for (int e = 0; e < NEXP; ++e) acc[e] += xv * gw[d * NEXP + e];
    }
    #pragma unroll
    for (int e = 0; e < NEXP; ++e) red[e][tid] = acc[e];
    __syncthreads();
    for (int s = 64; s > 0; s >>= 1) {
        if (tid < s) {
            #pragma unroll
            for (int e = 0; e < NEXP; ++e) red[e][tid] += red[e][tid + s];
        }
        __syncthreads();
    }
    if (tid == 0) {
        float lg[NEXP], p[NEXP];
        float m = -1e30f;
        #pragma unroll
        for (int e = 0; e < NEXP; ++e) { lg[e] = red[e][0]; m = fmaxf(m, lg[e]); }
        float sum = 0.f;
        #pragma unroll
        for (int e = 0; e < NEXP; ++e) { p[e] = __expf(lg[e] - m); sum += p[e]; }
        int best = 0;
        #pragma unroll
        for (int e = 0; e < NEXP; ++e) {
            p[e] /= sum;
            probs[t * NEXP + e] = p[e];
        }
        #pragma unroll
        for (int e = 1; e < NEXP; ++e) if (p[e] > p[best]) best = e;
        gate[t] = p[best];
        expert[t] = best;
        atomicAdd(&counts[best], 1);
    }
}

__global__ void zero_counts_kernel(int* counts) {
    if (threadIdx.x < NEXP) counts[threadIdx.x] = 0;
}

__global__ void offsets_kernel(const int* __restrict__ counts,
                               int* __restrict__ offsets, int* __restrict__ cursors) {
    if (threadIdx.x == 0) {
        int acc = 0;
        for (int e = 0; e < NEXP; ++e) { offsets[e] = acc; cursors[e] = acc; acc += counts[e]; }
    }
}

__global__ __launch_bounds__(256)
void scatter_kernel(const int* __restrict__ expert, int* __restrict__ cursors,
                    int* __restrict__ perm) {
    int t = blockIdx.x * 256 + threadIdx.x;
    if (t < TOK) {
        int e = expert[t];
        int pos = atomicAdd(&cursors[e], 1);
        perm[pos] = t;
    }
}

// Deterministic aux-loss: aux = NEXP * sum_e (count_e/TOK) * (mean_t probs[t,e])
__global__ __launch_bounds__(256)
void aux_kernel(const float* __restrict__ probs, const int* __restrict__ counts,
                float* __restrict__ aux_out)
{
    __shared__ float red[NEXP][256];
    const int tid = threadIdx.x;
    float part[NEXP] = {0.f, 0.f, 0.f, 0.f};
    for (int t = tid; t < TOK; t += 256) {
        #pragma unroll
        for (int e = 0; e < NEXP; ++e) part[e] += probs[t * NEXP + e];
    }
    #pragma unroll
    for (int e = 0; e < NEXP; ++e) red[e][tid] = part[e];
    __syncthreads();
    for (int s = 128; s > 0; s >>= 1) {
        if (tid < s) {
            #pragma unroll
            for (int e = 0; e < NEXP; ++e) red[e][tid] += red[e][tid + s];
        }
        __syncthreads();
    }
    if (tid == 0) {
        float aux = 0.f;
        #pragma unroll
        for (int e = 0; e < NEXP; ++e) {
            float f = (float)counts[e] / (float)TOK;
            float p = red[e][0] / (float)TOK;
            aux += f * p;
        }
        aux_out[0] = (float)NEXP * aux;
    }
}

// ---------------------------------------------------------------------------
// Host-side orchestration
// ---------------------------------------------------------------------------
extern "C" void kernel_launch(void* const* d_in, const int* in_sizes, int n_in,
                              void* d_out, int out_size, void* d_ws, size_t ws_size,
                              hipStream_t stream)
{
    const float* x         = (const float*)d_in[0];
    const float* norm1_w   = (const float*)d_in[1];
    const float* norm2_w   = (const float*)d_in[2];
    const float* in_proj_w = (const float*)d_in[3];
    const float* conv_w    = (const float*)d_in[4];
    const float* conv_b    = (const float*)d_in[5];
    const float* x_proj_w  = (const float*)d_in[6];
    const float* dt_proj_w = (const float*)d_in[7];
    const float* dt_proj_b = (const float*)d_in[8];
    const float* A_log     = (const float*)d_in[9];
    const float* D_param   = (const float*)d_in[10];
    const float* out_proj_w= (const float*)d_in[11];
    const float* ls_ssm_g  = (const float*)d_in[12];
    const float* ls_ffn_g  = (const float*)d_in[13];
    const float* gate_w    = (const float*)d_in[14];
    const float* w1        = (const float*)d_in[15];
    const float* w3        = (const float*)d_in[16];
    const float* w2        = (const float*)d_in[17];

    float* out = (float*)d_out;                        // [TOK, DMODEL]
    float* aux = out + (long)TOK * DMODEL;             // scalar

    // workspace layout (floats); total ~165 MB
    float* ws     = (float*)d_ws;
    float* xn     = ws;                                // 4096*768
    float* xz     = xn     + (long)TOK * DMODEL;       // 4096*3072 (reused as hbuf)
    float* xc     = xz     + (long)TOK * 2 * DINNER;   // 4096*1536
    float* dbc    = xc     + (long)TOK * DINNER;       // 4096*80
    float* delta  = dbc    + (long)TOK * 80;           // 4096*1536
    float* ygated = delta  + (long)TOK * DINNER;       // 4096*1536
    float* hbufR  = xz;                                // reuse xz after scan
    float* hres   = ygated + (long)TOK * DINNER;       // 4096*768
    float* hn     = hres   + (long)TOK * DMODEL;       // 4096*768
    float* probs  = hn     + (long)TOK * DMODEL;       // 4096*4
    float* gate   = probs  + (long)TOK * NEXP;         // 4096
    int*   ibase  = (int*)(gate + TOK);
    int*   expert = ibase;                             // 4096
    int*   perm   = expert + TOK;                      // 4096
    int*   counts = perm + TOK;                        // 4
    int*   offs   = counts + NEXP;                     // 4
    int*   curs   = offs + NEXP;                       // 4

    const dim3 blk128(128), blk256(256);

    // 1) rmsnorm1
    rmsnorm_kernel<<<TOK, blk256, 0, stream>>>(x, norm1_w, xn);

    // 2) xz = xn @ in_proj_w   [4096 x 768 x 3072]
    wmma_gemm_kernel<<<dim3(2 * DINNER / 64, TOK / 32, 1), blk128, 0, stream>>>(
        xn, in_proj_w, nullptr, xz, TOK, 2 * DINNER, DMODEL,
        DMODEL, 2 * DINNER, 2 * DINNER,
        nullptr, nullptr, nullptr, 0, 0, 0L,
        nullptr, 0, nullptr, 0, nullptr, nullptr);

    // 3) xc = silu(depthwise_conv(xs) + conv_b)
    conv_silu_kernel<<<(TOK * DINNER) / 256, blk256, 0, stream>>>(xz, conv_w, conv_b, xc);

    // 4) dbc = xc @ x_proj_w   [4096 x 1536 x 80]
    wmma_gemm_kernel<<<dim3(2, TOK / 32, 1), blk128, 0, stream>>>(
        xc, x_proj_w, nullptr, dbc, TOK, 80, DINNER,
        DINNER, 80, 80,
        nullptr, nullptr, nullptr, 0, 0, 0L,
        nullptr, 0, nullptr, 0, nullptr, nullptr);

    // 5) delta = softplus(dt @ dt_proj_w + dt_proj_b)   [4096 x 48 x 1536]
    wmma_gemm_kernel<<<dim3(DINNER / 64, TOK / 32, 1), blk128, 0, stream>>>(
        dbc, dt_proj_w, nullptr, delta, TOK, DINNER, DTRANK,
        80, DINNER, DINNER,
        nullptr, nullptr, nullptr, 0, 0, 0L,
        dt_proj_b, 1, nullptr, 0, nullptr, nullptr);

    // 6) selective scan -> ygated
    scan_kernel<<<dim3(DINNER / 256, BB), blk256, 0, stream>>>(
        delta, dbc, xc, xz, A_log, D_param, ygated);

    // 7) h = x + ls_ssm_g * (ygated @ out_proj_w)   [4096 x 1536 x 768]
    wmma_gemm_kernel<<<dim3(DMODEL / 64, TOK / 32, 1), blk128, 0, stream>>>(
        ygated, out_proj_w, nullptr, hres, TOK, DMODEL, DINNER,
        DINNER, DMODEL, DMODEL,
        nullptr, nullptr, nullptr, 0, 0, 0L,
        nullptr, 0, x, DMODEL, ls_ssm_g, nullptr);

    // 8) rmsnorm2
    rmsnorm_kernel<<<TOK, blk256, 0, stream>>>(hres, norm2_w, hn);

    // 9) router + bucketing
    zero_counts_kernel<<<1, 32, 0, stream>>>(counts);
    router_kernel<<<TOK, blk128, 0, stream>>>(hn, gate_w, probs, gate, expert, counts);
    offsets_kernel<<<1, 32, 0, stream>>>(counts, offs, curs);
    scatter_kernel<<<TOK / 256, blk256, 0, stream>>>(expert, curs, perm);

    // 10) hbuf = silu(hn_g @ w1[e]) * (hn_g @ w3[e])  (gathered rows, per expert)
    wmma_gemm_kernel<<<dim3(DFF / 64, TOK / 32, NEXP), blk128, 0, stream>>>(
        hn, w1, w3, hbufR, TOK, DFF, DMODEL,
        DMODEL, DFF, DFF,
        counts, offs, perm, /*gatherA=*/1, /*scatterC=*/0,
        (long)DMODEL * DFF,
        nullptr, 2, nullptr, 0, nullptr, nullptr);

    // 11) out = h + ls_ffn_g * gate[t] * (hbuf @ w2[e])  (scatter rows to tokens)
    wmma_gemm_kernel<<<dim3(DMODEL / 64, TOK / 32, NEXP), blk128, 0, stream>>>(
        hbufR, w2, nullptr, out, TOK, DMODEL, DFF,
        DFF, DMODEL, DMODEL,
        counts, offs, perm, /*gatherA=*/0, /*scatterC=*/1,
        (long)DFF * DMODEL,
        nullptr, 0, hres, DMODEL, ls_ffn_g, gate);

    // 12) aux loss (deterministic tree reduction)
    aux_kernel<<<1, blk256, 0, stream>>>(probs, counts, aux);
}
